// LatSim_38354057954023
// MI455X (gfx1250) — matrix operand order repaired and checked by reading, not compiled, not run on hardware
//
#include <hip/hip_runtime.h>

typedef __attribute__((ext_vector_type(16))) _Float16 v16h;
typedef __attribute__((ext_vector_type(8)))  _Float16 v8h;
typedef __attribute__((ext_vector_type(8)))  float    v8f;

constexpr int kD      = 1024;
constexpr int kLD     = 64;
constexpr int kNTR    = 8192;
constexpr int kNT     = 8192;
constexpr int kDY     = 128;
constexpr int kSplit  = 4;
constexpr int kKChunk = kNTR / kSplit;    // 2048 keys per block
constexpr int kKStr   = 72;               // padded LDS row stride for K (halves)
constexpr int kVStr   = 40;               // padded LDS row stride for V^T (halves)

#define WMMA_F16(A, B, C) \
  __builtin_amdgcn_wmma_f32_16x16x32_f16(false, (A), false, (B), (short)0, (C), false, false)

// ---- CDNA5 async global->LDS copy (b128), tracked by ASYNCcnt --------------
__device__ __forceinline__ unsigned lds_off(const void* p) {
  return (unsigned)(uintptr_t)(__attribute__((address_space(3))) const void*)p;
}
__device__ __forceinline__ void async_b128(const _Float16* g, _Float16* l) {
  asm volatile("global_load_async_to_lds_b128 %0, %1, off"
               :: "v"(lds_off(l)), "v"((unsigned long long)(uintptr_t)g)
               : "memory");
}
__device__ __forceinline__ void wait_async0() {
  asm volatile("s_wait_asynccnt 0x0" ::: "memory");
}

// ---------------------------------------------------------------- utilities
__global__ void zero_f32(float* __restrict__ p, int n) {
  int i = blockIdx.x * blockDim.x + threadIdx.x;
  if (i < n) p[i] = 0.0f;
}

__global__ void col_sum(const float* __restrict__ x, float* __restrict__ muSum) {
  int col = blockIdx.x * blockDim.x + threadIdx.x;
  int r0  = blockIdx.y * 256;
  float s = 0.0f;
#pragma unroll 4
  for (int r = 0; r < 256; ++r) s += x[(size_t)(r0 + r) * kD + col];
  atomicAdd(&muSum[col], s);
}

// dst[c*R + r] = (f16) src[r*C + c]
__global__ void transpose_f32_to_f16(const float* __restrict__ src,
                                     _Float16* __restrict__ dst, int R, int C) {
  int i = blockIdx.x * blockDim.x + threadIdx.x;
  int c = i / R, r = i % R;
  dst[i] = (_Float16)src[(size_t)r * C + c];
}

// ---------------------------------------------------------------- projection
__global__ void __launch_bounds__(256)
proj_kernel(const float* __restrict__ x, const float* __restrict__ muSum,
            const _Float16* __restrict__ At, _Float16* __restrict__ qh) {
  const int lane = threadIdx.x & 31;
  const int wave = threadIdx.x >> 5;
  const int lq   = lane & 15;
  const int half = lane >> 4;
  const int rowBase = (blockIdx.x * 8 + wave) * 16;
  const float invN = 1.0f / (float)kNTR;

  const float* xrow = x + (size_t)(rowBase + lq) * kD + half * 16;
  const float* mrow = muSum + half * 16;

  v8f acc0 = {}, acc1 = {}, acc2 = {}, acc3 = {};
#pragma unroll 1
  for (int k0 = 0; k0 < kD; k0 += 32) {
    v8f xc0 = *(const v8f*)(xrow + k0);
    v8f xc1 = *(const v8f*)(xrow + k0 + 8);
    v8f m0  = *(const v8f*)(mrow + k0);
    v8f m1  = *(const v8f*)(mrow + k0 + 8);
    v16h a;
#pragma unroll
    for (int i = 0; i < 8; ++i) {
      a[i]     = (_Float16)(xc0[i] - m0[i] * invN);
      a[i + 8] = (_Float16)(xc1[i] - m1[i] * invN);
    }
    const _Float16* bp = At + (size_t)lq * kD + k0 + half * 16;
    v16h b0 = *(const v16h*)(bp);
    v16h b1 = *(const v16h*)(bp + 16 * kD);
    v16h b2 = *(const v16h*)(bp + 32 * kD);
    v16h b3 = *(const v16h*)(bp + 48 * kD);
    acc0 = WMMA_F16(a, b0, acc0);
    acc1 = WMMA_F16(a, b1, acc1);
    acc2 = WMMA_F16(a, b2, acc2);
    acc3 = WMMA_F16(a, b3, acc3);
  }
#pragma unroll
  for (int r = 0; r < 8; ++r) {
    int row = rowBase + half * 8 + r;
    _Float16* op = qh + (size_t)row * kLD + lq;
    op[0]  = (_Float16)acc0[r];
    op[16] = (_Float16)acc1[r];
    op[32] = (_Float16)acc2[r];
    op[48] = (_Float16)acc3[r];
  }
}

// ---------------------------------------------------------------- attention
// Block = 8 waves x 16 query rows = 128 queries; key range split 4-way across
// blockIdx.y. K/V tiles staged in LDS via double-buffered async copies and
// shared by all 8 waves (8x L2 traffic reduction). Wave computes S^T = K*Q^T
// so exp'd scores land directly in the P@V A-operand layout.
__global__ void __launch_bounds__(256)
attn_kernel(const _Float16* __restrict__ Qh, const _Float16* __restrict__ Kh,
            const _Float16* __restrict__ Vt, float* __restrict__ Op,
            float* __restrict__ Mp, float* __restrict__ Sp) {
  __shared__ _Float16 lK[2][32 * kKStr];    //  9.2 KB
  __shared__ _Float16 lV[2][128 * kVStr];   // 20.5 KB

  const int tid  = threadIdx.x;
  const int lane = tid & 31;
  const int wave = tid >> 5;
  const int lq   = lane & 15;
  const int half = lane >> 4;
  const int qBase = (blockIdx.x * 8 + wave) * 16;
  const int split = blockIdx.y;
  const int kt0   = split * kKChunk;
  const int srcl  = half * 8;
  const float LOG2E = 1.44269504088896340736f;

  // staging assignments (per thread, one b128 for K, two for V)
  const int krow = tid >> 3, kcol = (tid & 7) * 8;
  const int vdy  = tid >> 1, vpart = (tid & 1) * 16;
  const _Float16* gK = Kh + (size_t)(kt0 + krow) * kLD + kcol;
  const _Float16* gV = Vt + (size_t)vdy * kNTR + kt0 + vpart;

  // prologue: fill buffer 0
  async_b128(gK, &lK[0][krow * kKStr + kcol]);
  async_b128(gV,      &lV[0][vdy * kVStr + vpart]);
  async_b128(gV + 8,  &lV[0][vdy * kVStr + vpart + 8]);

  // Q as B-operand of K*Q^T (kept in VGPRs for the whole key loop)
  const _Float16* qp = Qh + (size_t)(qBase + lq) * kLD + half * 16;
  v16h bq0 = *(const v16h*)(qp);
  v16h bq1 = *(const v16h*)(qp + 32);

  v8f o[8] = {{}, {}, {}, {}, {}, {}, {}, {}};
  float m = -1e30f, s = 0.0f;

  wait_async0();
  __syncthreads();

#pragma unroll 1
  for (int it = 0; it < kKChunk / 32; ++it) {
    const int cur = it & 1;
    // prefetch next tile into the other buffer (read last at barrier it-1)
    if (it + 1 < kKChunk / 32) {
      const _Float16* nK = gK + (size_t)(it + 1) * 32 * kLD;
      const _Float16* nV = gV + (it + 1) * 32;
      async_b128(nK,     &lK[cur ^ 1][krow * kKStr + kcol]);
      async_b128(nV,     &lV[cur ^ 1][vdy * kVStr + vpart]);
      async_b128(nV + 8, &lV[cur ^ 1][vdy * kVStr + vpart + 8]);
    }

    // K tile as A-operand from LDS (padded rows -> conflict-free b128 reads)
    const _Float16* kp0 = &lK[cur][lq * kKStr + half * 16];
    const _Float16* kp1 = kp0 + 16 * kKStr;
    v8h c0 = *(const v8h*)(kp0);
    v8h c1 = *(const v8h*)(kp0 + 8);
    v8h c2 = *(const v8h*)(kp0 + 32);
    v8h c3 = *(const v8h*)(kp0 + 40);
    v8h e0 = *(const v8h*)(kp1);
    v8h e1 = *(const v8h*)(kp1 + 8);
    v8h e2 = *(const v8h*)(kp1 + 32);
    v8h e3 = *(const v8h*)(kp1 + 40);
    v16h ak0 = __builtin_shufflevector(c0, c1, 0,1,2,3,4,5,6,7,8,9,10,11,12,13,14,15);
    v16h ak1 = __builtin_shufflevector(c2, c3, 0,1,2,3,4,5,6,7,8,9,10,11,12,13,14,15);
    v16h al0 = __builtin_shufflevector(e0, e1, 0,1,2,3,4,5,6,7,8,9,10,11,12,13,14,15);
    v16h al1 = __builtin_shufflevector(e2, e3, 0,1,2,3,4,5,6,7,8,9,10,11,12,13,14,15);

    v8f t0 = {}, t1 = {};
    t0 = WMMA_F16(ak0, bq0, t0);
    t0 = WMMA_F16(ak1, bq1, t0);
    t1 = WMMA_F16(al0, bq0, t1);
    t1 = WMMA_F16(al1, bq1, t1);

    // online softmax in log2 domain
    float p0[8], p1[8];
    float tmax = -1e30f;
#pragma unroll
    for (int r = 0; r < 8; ++r) {
      p0[r] = t0[r] * LOG2E;
      p1[r] = t1[r] * LOG2E;
      tmax = fmaxf(tmax, fmaxf(p0[r], p1[r]));
    }
    tmax = fmaxf(tmax, __shfl_xor(tmax, 16));
    float mnew  = fmaxf(m, tmax);
    float alpha = exp2f(m - mnew);
    float rs = 0.0f;
    v16h ap;
#pragma unroll
    for (int r = 0; r < 8; ++r) {
      float w0 = exp2f(p0[r] - mnew);
      float w1 = exp2f(p1[r] - mnew);
      rs += w0 + w1;
      ap[r]     = (_Float16)w0;
      ap[8 + r] = (_Float16)w1;
    }
    rs += __shfl_xor(rs, 16);
    s = s * alpha + rs;
    m = mnew;

    float aO[8];
#pragma unroll
    for (int r = 0; r < 8; ++r) aO[r] = __shfl(alpha, srcl + r);
#pragma unroll
    for (int j = 0; j < 8; ++j)
#pragma unroll
      for (int r = 0; r < 8; ++r) o[j][r] *= aO[r];

    // O += P * V from the LDS V^T tile
    const _Float16* vp = &lV[cur][lq * kVStr + half * 8];
#pragma unroll
    for (int j = 0; j < 8; ++j) {
      v8h v0 = *(const v8h*)(vp + j * 16 * kVStr);
      v8h v1 = *(const v8h*)(vp + j * 16 * kVStr + 16);
      v16h bv = __builtin_shufflevector(v0, v1, 0,1,2,3,4,5,6,7,8,9,10,11,12,13,14,15);
      o[j] = WMMA_F16(ap, bv, o[j]);
    }

    wait_async0();      // next-tile async copies landed
    __syncthreads();    // everyone done reading cur before it is overwritten
  }

  // unnormalized partials for the split-K combine
  if (half == 0) {
    Mp[(size_t)split * kNT + qBase + lq] = m;
    Sp[(size_t)split * kNT + qBase + lq] = s;
  }
#pragma unroll
  for (int r = 0; r < 8; ++r) {
    float* op = Op + (size_t)split * kNT * kDY + (size_t)(qBase + half * 8 + r) * kDY + lq;
#pragma unroll
    for (int j = 0; j < 8; ++j) op[j * 16] = o[j][r];
  }
}

// merge the 4 key-range splits: out = sum_i 2^(m_i-M) O_i / sum_i 2^(m_i-M) s_i
__global__ void combine_kernel(const float* __restrict__ Op, const float* __restrict__ Mp,
                               const float* __restrict__ Sp, float* __restrict__ out) {
  int i = blockIdx.x * blockDim.x + threadIdx.x;   // over NT*DY
  int q = i >> 7;
  float M = Mp[q];
#pragma unroll
  for (int sp = 1; sp < kSplit; ++sp) M = fmaxf(M, Mp[sp * kNT + q]);
  float S = 0.0f, O = 0.0f;
#pragma unroll
  for (int sp = 0; sp < kSplit; ++sp) {
    float w = exp2f(Mp[sp * kNT + q] - M);
    S += Sp[sp * kNT + q] * w;
    O += Op[(size_t)sp * kNT * kDY + i] * w;
  }
  out[i] = O / S;
}

// ---------------------------------------------------------------- launch
extern "C" void kernel_launch(void* const* d_in, const int* in_sizes, int n_in,
                              void* d_out, int out_size, void* d_ws, size_t ws_size,
                              hipStream_t stream) {
  const float* xtr = (const float*)d_in[0];   // [8192,1024]
  const float* ytr = (const float*)d_in[1];   // [8192,128]
  const float* xt  = (const float*)d_in[2];   // [8192,1024]
  const float* A   = (const float*)d_in[3];   // [1024,64]
  float* out = (float*)d_out;                 // [8192,128]

  char* ws = (char*)d_ws;
  size_t off = 0;
  float*    muSum = (float*)(ws + off); off += (size_t)kD * 4;                 // 4 KB
  _Float16* At    = (_Float16*)(ws + off); off += (size_t)kLD * kD * 2;        // 128 KB
  _Float16* Qh    = (_Float16*)(ws + off); off += (size_t)kNT * kLD * 2;       // 1 MB
  _Float16* Kh    = (_Float16*)(ws + off); off += (size_t)kNTR * kLD * 2;      // 1 MB
  _Float16* Vt    = (_Float16*)(ws + off); off += (size_t)kDY * kNTR * 2;      // 2 MB
  float*    Mp    = (float*)(ws + off); off += (size_t)kSplit * kNT * 4;
  float*    Sp    = (float*)(ws + off); off += (size_t)kSplit * kNT * 4;
  float*    Op    = (float*)(ws + off); off += (size_t)kSplit * kNT * kDY * 4; // 16 MB

  zero_f32<<<(kD + 255) / 256, 256, 0, stream>>>(muSum, kD);
  col_sum<<<dim3(kD / 256, kNTR / 256), 256, 0, stream>>>(xtr, muSum);
  transpose_f32_to_f16<<<(kD * kLD) / 256, 256, 0, stream>>>(A, At, kD, kLD);
  transpose_f32_to_f16<<<(kNTR * kDY) / 256, 256, 0, stream>>>(ytr, Vt, kNTR, kDY);
  proj_kernel<<<kNT / 128, 256, 0, stream>>>(xt, muSum, At, Qh);
  proj_kernel<<<kNTR / 128, 256, 0, stream>>>(xtr, muSum, At, Kh);
  attn_kernel<<<dim3(kNT / 128, kSplit), 256, 0, stream>>>(Qh, Kh, Vt, Op, Mp, Sp);
  combine_kernel<<<(kNT * kDY) / 256, 256, 0, stream>>>(Op, Mp, Sp, out);
}
